// BinarizedVGG19_13924283974418
// MI455X (gfx1250) — 8-bit matrix operand order repaired and checked by reading, compiled, not run on hardware
//
#include <hip/hip_runtime.h>
#include <hip/hip_bf16.h>
#include <stdint.h>

typedef __attribute__((ext_vector_type(8))) int v8i;

#define BATCH 8

__device__ __forceinline__ void bvgg_async_ld_b128(unsigned lds_off, const void* g) {
  asm volatile("global_load_async_to_lds_b128 %0, %1, off"
               :: "v"(lds_off), "v"(g) : "memory");
}
__device__ __forceinline__ void bvgg_wait_async0() {
  asm volatile("s_wait_asynccnt 0" ::: "memory");
}

__global__ void bvgg_binarize_input_kernel(const float* __restrict__ x,
                                           char* __restrict__ y, int n) {
  int i = blockIdx.x * blockDim.x + threadIdx.x;
  if (i < n) y[i] = (x[i] >= 0.0f) ? (char)1 : (char)-1;
}

__global__ void bvgg_pack_weights_kernel(const float* __restrict__ w,
                                         unsigned int* __restrict__ frag,
                                         int K, int Cout, int Ktiles, int Ntiles) {
  int idx = blockIdx.x * blockDim.x + threadIdx.x;
  int total = Ktiles * Ntiles * 32 * 8;
  if (idx >= total) return;
  int v    = idx & 7;
  int lane = (idx >> 3) & 31;
  int nt   = (idx >> 8) % Ntiles;
  int kt   = (idx >> 8) / Ntiles;
  int half = lane >> 4;
  int n    = nt * 16 + (lane & 15);
  int kbase = kt * 64 + (v >> 1) * 16 + half * 8 + (v & 1) * 4;
  unsigned int word = 0u;
  for (int byte = 0; byte < 4; ++byte) {
    int kk = kbase + byte;
    if (kk < K) {
      float wv = w[(size_t)kk * Cout + n];
      unsigned char s = (wv >= 0.0f) ? 0x01u : 0xFFu;
      word |= ((unsigned int)s) << (8 * byte);
    }
  }
  frag[idx] = word;
}

template <int CIN, int STOREF>
__global__ void bvgg_conv_wmma_kernel(const char* __restrict__ act,
                                      const unsigned int* __restrict__ wfrag,
                                      const float* __restrict__ bias,
                                      void* __restrict__ outp,
                                      int H, int W, int Cout) {
  constexpr int K = 9 * CIN;
  constexpr int Ktiles = (K + 63) / 64;
  constexpr int QT = (CIN >= 64) ? (CIN / 64) : 1;

  __shared__ __align__(16) unsigned char bvgg_smem[2 * 4096];
  const unsigned smbase = (unsigned)(size_t)&bvgg_smem[0];

  const int M = BATCH * H * W;
  const int Mtiles = M >> 4;
  const int Ntiles = Cout >> 4;

  const int wave = threadIdx.x >> 5;
  const int lane = threadIdx.x & 31;
  const int half = lane >> 4;
  const int l15  = lane & 15;

  const int mt_raw = blockIdx.x * 8 + wave;
  const bool wave_valid = (mt_raw < Mtiles);
  const int mt  = wave_valid ? mt_raw : (Mtiles - 1);
  const int nt0 = blockIdx.y * 4;

  int p  = mt * 16 + l15;
  int xw = p % W;
  int t1 = p / W;
  int yh = t1 % H;
  int bb = t1 / H;

  v8i c0 = {0,0,0,0,0,0,0,0}, c1 = c0, c2 = c0, c3 = c0;

  const char* wfb = (const char*)wfrag;
  auto issue = [&](int k) {
    const void* g = wfb + ((size_t)k * Ntiles + nt0) * 1024
                        + (size_t)threadIdx.x * 16;
    unsigned l = smbase + ((unsigned)(k & 1)) * 4096u
                        + (unsigned)threadIdx.x * 16u;
    bvgg_async_ld_b128(l, g);
  };
  auto load_b = [&](int kt, int j) -> v8i {
    return *(const v8i*)&bvgg_smem[(kt & 1) * 4096 + j * 1024 + lane * 32];
  };

  issue(0);

  if constexpr (CIN == 3) {
    bvgg_wait_async0();
    __syncthreads();
    v8i a;
#pragma unroll
    for (int v = 0; v < 8; ++v) {
      int kb = (v >> 1) * 16 + half * 8 + (v & 1) * 4;
      unsigned int word = 0u;
#pragma unroll
      for (int byte = 0; byte < 4; ++byte) {
        int kk = kb + byte;
        if (kk < K) {
          int t  = kk / 3;
          int cc = kk - t * 3;
          int yy = yh + (t / 3) - 1;
          int xx = xw + (t % 3) - 1;
          if (yy >= 0 && yy < H && xx >= 0 && xx < W) {
            unsigned char bv = (unsigned char)
                act[(((size_t)bb * H + yy) * W + xx) * CIN + cc];
            word |= ((unsigned int)bv) << (8 * byte);
          }
        }
      }
      a[v] = (int)word;
    }
    v8i b0 = load_b(0, 0), b1 = load_b(0, 1), b2 = load_b(0, 2), b3 = load_b(0, 3);
    c0 = __builtin_amdgcn_wmma_i32_16x16x64_iu8(true, a, true, b0, c0, false, false);
    c1 = __builtin_amdgcn_wmma_i32_16x16x64_iu8(true, a, true, b1, c1, false, false);
    c2 = __builtin_amdgcn_wmma_i32_16x16x64_iu8(true, a, true, b2, c2, false, false);
    c3 = __builtin_amdgcn_wmma_i32_16x16x64_iu8(true, a, true, b3, c3, false, false);
  } else {
    int kt = 0;
#pragma unroll
    for (int t = 0; t < 9; ++t) {
      const int dy = t / 3 - 1, dx = t % 3 - 1;
      const int yy = yh + dy, xx = xw + dx;
      const bool valid = (yy >= 0 && yy < H && xx >= 0 && xx < W);
      const char* hp = act + (((size_t)bb * H + yy) * W + xx) * CIN + half * 8;
      if (valid) __builtin_prefetch(hp, 0, 0);
      for (int q = 0; q < QT; ++q, ++kt) {
        bvgg_wait_async0();
        __syncthreads();
        if (kt + 1 < Ktiles) issue(kt + 1);
        v8i b0 = load_b(kt, 0), b1 = load_b(kt, 1);
        v8i b2 = load_b(kt, 2), b3 = load_b(kt, 3);
        v8i a;
#pragma unroll
        for (int v = 0; v < 8; ++v) {
          const int off = q * 64 + (v >> 1) * 16 + (v & 1) * 4;
          unsigned int word = 0u;
          if (valid) word = *(const unsigned int*)(hp + off);
          a[v] = (int)word;
        }
        c0 = __builtin_amdgcn_wmma_i32_16x16x64_iu8(true, a, true, b0, c0, false, false);
        c1 = __builtin_amdgcn_wmma_i32_16x16x64_iu8(true, a, true, b1, c1, false, false);
        c2 = __builtin_amdgcn_wmma_i32_16x16x64_iu8(true, a, true, b2, c2, false, false);
        c3 = __builtin_amdgcn_wmma_i32_16x16x64_iu8(true, a, true, b3, c3, false, false);
      }
    }
  }

  if (wave_valid) {
#pragma unroll
    for (int j = 0; j < 4; ++j) {
      v8i c = (j == 0) ? c0 : (j == 1) ? c1 : (j == 2) ? c2 : c3;
      int n = (nt0 + j) * 16 + l15;
      float bv = bias[n];
#pragma unroll
      for (int r = 0; r < 8; ++r) {
        int m = mt * 16 + half * 8 + r;
        float f = (float)c[r] + bv;
        f = fmaxf(f, 0.0f);
        asm volatile("" : "+v"(f));
        if constexpr (STOREF) {
          ((float*)outp)[(size_t)m * Cout + n] = f;
        } else {
          ((char*)outp)[(size_t)m * Cout + n] = (f >= 0.0f) ? (char)1 : (char)-1;
        }
      }
    }
  }
}

__global__ void bvgg_maxpool_i8_kernel(const char* __restrict__ in,
                                       char* __restrict__ out,
                                       int Hout, int Wout, int C) {
  int i = blockIdx.x * blockDim.x + threadIdx.x;
  int total = BATCH * Hout * Wout * C;
  if (i >= total) return;
  int cc = i % C;
  int t  = i / C;
  int xo = t % Wout; t /= Wout;
  int yo = t % Hout; t /= Hout;
  int b  = t;
  int Win = Wout * 2, Hin = Hout * 2;
  size_t base = (((size_t)b * Hin + yo * 2) * Win + xo * 2) * C + cc;
  char m0 = in[base];
  char m1 = in[base + C];
  char m2 = in[base + (size_t)Win * C];
  char m3 = in[base + (size_t)Win * C + C];
  char ma = (m0 > m1) ? m0 : m1;
  char mb = (m2 > m3) ? m2 : m3;
  out[i] = (ma > mb) ? ma : mb;
}

__global__ void bvgg_maxpool_f32_kernel(const float* __restrict__ in,
                                        float* __restrict__ out,
                                        int Hout, int Wout, int C) {
  int i = blockIdx.x * blockDim.x + threadIdx.x;
  int total = BATCH * Hout * Wout * C;
  if (i >= total) return;
  int cc = i % C;
  int t  = i / C;
  int xo = t % Wout; t /= Wout;
  int yo = t % Hout; t /= Hout;
  int b  = t;
  int Win = Wout * 2, Hin = Hout * 2;
  size_t base = (((size_t)b * Hin + yo * 2) * Win + xo * 2) * C + cc;
  float m0 = in[base];
  float m1 = in[base + C];
  float m2 = in[base + (size_t)Win * C];
  float m3 = in[base + (size_t)Win * C + C];
  out[i] = fmaxf(fmaxf(m0, m1), fmaxf(m2, m3));
}

static void bvgg_launch_conv(int cin, int storeF, const char* act,
                             const unsigned int* wf, const float* bias,
                             void* outp, int H, int W, int Cout,
                             hipStream_t s) {
  int Mtiles = (BATCH * H * W) / 16;
  int gx = (Mtiles + 7) / 8;
  int gy = (Cout / 16) / 4;
  dim3 g(gx, gy), b(256);
  if (storeF) {
    bvgg_conv_wmma_kernel<512, 1><<<g, b, 0, s>>>(act, wf, bias, outp, H, W, Cout);
    return;
  }
  switch (cin) {
    case 3:   bvgg_conv_wmma_kernel<3,   0><<<g, b, 0, s>>>(act, wf, bias, outp, H, W, Cout); break;
    case 64:  bvgg_conv_wmma_kernel<64,  0><<<g, b, 0, s>>>(act, wf, bias, outp, H, W, Cout); break;
    case 128: bvgg_conv_wmma_kernel<128, 0><<<g, b, 0, s>>>(act, wf, bias, outp, H, W, Cout); break;
    case 256: bvgg_conv_wmma_kernel<256, 0><<<g, b, 0, s>>>(act, wf, bias, outp, H, W, Cout); break;
    case 512: bvgg_conv_wmma_kernel<512, 0><<<g, b, 0, s>>>(act, wf, bias, outp, H, W, Cout); break;
    default: break;
  }
}

extern "C" void kernel_launch(void* const* d_in, const int* in_sizes, int n_in,
                              void* d_out, int out_size, void* d_ws, size_t ws_size,
                              hipStream_t stream) {
  (void)in_sizes; (void)n_in; (void)out_size; (void)ws_size;

  static const int CIN[16]  = {3, 64, 64, 128, 128, 256, 256, 256,
                               256, 512, 512, 512, 512, 512, 512, 512};
  static const int COUT[16] = {64, 64, 128, 128, 256, 256, 256, 256,
                               512, 512, 512, 512, 512, 512, 512, 512};

  const float* inp = (const float*)d_in[0];
  const float* wsrc[16];
  const float* bsrc[16];
  for (int i = 0; i < 16; ++i) {
    wsrc[i] = (const float*)d_in[1 + i];
    bsrc[i] = (const float*)d_in[17 + i];
  }

  char* base = (char*)d_ws;
  const size_t ACT = (size_t)BATCH * 224 * 224 * 64;
  char* actA = base;
  char* actB = base + ACT;
  char* wf   = base + 2 * ACT;
  size_t wfoff[16];
  size_t woff = 0;
  for (int i = 0; i < 16; ++i) {
    int K  = 9 * CIN[i];
    int Kt = (K + 63) / 64;
    int Nt = COUT[i] / 16;
    wfoff[i] = woff;
    woff += (size_t)Kt * Nt * 1024;
  }
  float* ftmp = (float*)(wf + woff);

  {
    int n = BATCH * 224 * 224 * 3;
    bvgg_binarize_input_kernel<<<(n + 255) / 256, 256, 0, stream>>>(inp, actA, n);
  }

  for (int i = 0; i < 16; ++i) {
    int K  = 9 * CIN[i];
    int Kt = (K + 63) / 64;
    int Nt = COUT[i] / 16;
    int total = Kt * Nt * 256;
    bvgg_pack_weights_kernel<<<(total + 255) / 256, 256, 0, stream>>>(
        wsrc[i], (unsigned int*)(wf + wfoff[i]), K, COUT[i], Kt, Nt);
  }

  int H = 224, W = 224;
  char* cur = actA;
  char* nxt = actB;
  for (int i = 0; i < 16; ++i) {
    bool pool = (i == 1 || i == 3 || i == 7 || i == 11);
    if (i < 15) {
      bvgg_launch_conv(CIN[i], 0, cur, (const unsigned int*)(wf + wfoff[i]),
                       bsrc[i], (void*)nxt, H, W, COUT[i], stream);
      if (pool) {
        int Ho = H / 2, Wo = W / 2;
        int n = BATCH * Ho * Wo * COUT[i];
        bvgg_maxpool_i8_kernel<<<(n + 255) / 256, 256, 0, stream>>>(
            nxt, cur, Ho, Wo, COUT[i]);
        H = Ho; W = Wo;
      } else {
        char* t = cur; cur = nxt; nxt = t;
      }
    } else {
      bvgg_launch_conv(CIN[i], 1, cur, (const unsigned int*)(wf + wfoff[i]),
                       bsrc[i], (void*)ftmp, H, W, COUT[i], stream);
      int Ho = 7, Wo = 7;
      int n = BATCH * Ho * Wo * 512;
      bvgg_maxpool_f32_kernel<<<(n + 255) / 256, 256, 0, stream>>>(
          ftmp, (float*)d_out, Ho, Wo, 512);
    }
  }
}
